// GPS_15367392985091
// MI455X (gfx1250) — matrix operand order repaired and checked by reading, not compiled, not run on hardware
//
#include <hip/hip_runtime.h>

#define NN 65536
#define EE 524288
#define ENL (EE + NN)
#define CC 128
#define GG 256

typedef __attribute__((ext_vector_type(16))) _Float16 v16h;
typedef __attribute__((ext_vector_type(8)))  _Float16 v8h;
typedef __attribute__((ext_vector_type(8)))  float    v8f;

// ---- WMMA fragment helpers (CDNA5 16x16x32 f16 layouts, wave32) ----
// A (16x32, MxK): lanes 0-15 row M=lane, K group [hi*16, hi*16+16)
static __device__ __forceinline__ v16h load_a_frag(const _Float16* base, int ld,
                                                   int row0, int k0, int lane) {
  int hi = lane >> 4, r = lane & 15;
  const _Float16* p = base + (row0 + r) * ld + k0 + hi * 16;
  return *(const v16h*)p;
}
// B (32x16, KxN) from row-major [n][k] buffer: lane holds column n=lane&15,
// K group [hi*16, hi*16+16) -> one contiguous 32B LDS read per lane.
static __device__ __forceinline__ v16h load_b_frag(const _Float16* base, int ld,
                                                   int col0, int k0, int lane) {
  int hi = lane >> 4, r = lane & 15;
  return *(const v16h*)(base + (col0 + r) * ld + k0 + hi * 16);
}
static __device__ __forceinline__ v8f wmma16(v16h a, v16h b, v8f c) {
  return __builtin_amdgcn_wmma_f32_16x16x32_f16(false, a, false, b, (short)0, c,
                                                false, false);
}

// ---- generic WMMA GEMM: Y[N x Co] = X[N x Ci] @ W^T + bias (+resid), opt ReLU ----
// W is f16 row-major [Co x Ci]. Block = 128 thr (4 waves), tile 32x64, Ci<=256.
// Each wave computes a 16x32 slab: A-fragment reused across 2 WMMAs per k-step.
// W tile is a contiguous global range -> staged via global_load_async_to_lds_b128.
__global__ __launch_bounds__(128) void gemm_wmma(const float* __restrict__ X,
                                                 const _Float16* __restrict__ W,
                                                 const float* __restrict__ bias,
                                                 const float* __restrict__ resid,
                                                 float* __restrict__ Y,
                                                 int Ci, int Co, int relu) {
  __shared__ _Float16 sX[32 * 256];
  __shared__ _Float16 sW[64 * 256];
  int tid = threadIdx.x;
  int row0 = blockIdx.x * 32;
  int col0 = blockIdx.y * 64;
  // async copy of contiguous W tile (64*Ci halves) straight into LDS
  {
    const _Float16* Wp = W + (size_t)col0 * Ci;
    for (int i = tid * 8; i < 64 * Ci; i += 128 * 8) {
      unsigned lds = (unsigned)(uintptr_t)(&sW[i]);
      const void* ga = (const void*)(Wp + i);
      asm volatile("global_load_async_to_lds_b128 %0, %1, off"
                   :: "v"(lds), "v"(ga) : "memory");
    }
  }
  // X tile needs f32->f16 conversion, stage through VALU
  for (int i = tid; i < 32 * Ci; i += 128) {
    int r = i / Ci, k = i - r * Ci;
    sX[i] = (_Float16)X[(size_t)(row0 + r) * Ci + k];
  }
  asm volatile("s_wait_asynccnt 0" ::: "memory");
  __syncthreads();
  int wave = tid >> 5, lane = tid & 31;
  int wm = (wave >> 1) * 16;  // 0/16
  int wn = (wave & 1) * 32;   // 0/32
  v8f acc0 = {}, acc1 = {};
  for (int k0 = 0; k0 < Ci; k0 += 32) {
    v16h a  = load_a_frag(sX, Ci, wm, k0, lane);
    v16h b0 = load_b_frag(sW, Ci, wn, k0, lane);
    v16h b1 = load_b_frag(sW, Ci, wn + 16, k0, lane);
    acc0 = wmma16(a, b0, acc0);
    acc1 = wmma16(a, b1, acc1);
  }
  int hi = lane >> 4, r = lane & 15;
  int ca = col0 + wn + r;
  int cb = ca + 16;
  float bva = bias[ca], bvb = bias[cb];
#pragma unroll
  for (int j = 0; j < 8; ++j) {
    size_t row = (size_t)(row0 + wm + hi * 8 + j);
    float va = acc0[j] + bva;
    float vb = acc1[j] + bvb;
    if (relu) { va = fmaxf(va, 0.f); vb = fmaxf(vb, 0.f); }
    if (resid) { va += resid[row * Co + ca]; vb += resid[row * Co + cb]; }
    Y[row * Co + ca] = va;
    Y[row * Co + cb] = vb;
  }
}

// ---- node embedding: h = [x @ We^T + be | x_pe @ Wp^T + bp] ----
__global__ __launch_bounds__(256) void node_emb(const float* __restrict__ x,
                                                const float* __restrict__ xpe,
                                                const float* __restrict__ w,
                                                const float* __restrict__ b,
                                                const float* __restrict__ pw,
                                                const float* __restrict__ pb,
                                                float* __restrict__ h) {
  int n = blockIdx.x * 2 + (threadIdx.x >> 7);
  int c = threadIdx.x & 127;
  float v;
  if (c < 120) {
    const float* xr = x + (size_t)n * 32;
    const float* wr = w + c * 32;
    v = b[c];
#pragma unroll
    for (int k = 0; k < 32; ++k) v += xr[k] * wr[k];
  } else {
    int cc = c - 120;
    const float* xr = xpe + (size_t)n * 8;
    const float* wr = pw + cc * 8;
    v = pb[cc];
#pragma unroll
    for (int k = 0; k < 8; ++k) v += xr[k] * wr[k];
  }
  h[(size_t)n * 128 + c] = v;
}

// ---- GatedGCN edge stage: msg = Ax_i + sigmoid(Cx_i + Dx_j + Ee)*Bx_j ----
// ABCD layout [n][512] = [Ax|Bx|Cx|Dx]; Ee recomputed on the fly (16 MACs).
__global__ __launch_bounds__(256) void edge_gatedgcn(const float* __restrict__ ABCD,
                                                     const int* __restrict__ ei,
                                                     const float* __restrict__ ea,
                                                     const float* __restrict__ Ew,
                                                     const float* __restrict__ Eb,
                                                     float* __restrict__ agg) {
  int e = blockIdx.x * 2 + (threadIdx.x >> 7);
  if (e >= ENL) return;
  int c = threadIdx.x & 127;
  int s, d;
  if (e < EE) { s = ei[e]; d = ei[EE + e]; }
  else        { s = e - EE; d = s; }
  float Ee = Eb[c];
  if (e < EE) {
    const float* ap = ea + (size_t)e * 16;
    const float* wp = Ew + c * 16;
#pragma unroll
    for (int k = 0; k < 16; ++k) Ee += ap[k] * wp[k];
  }
  const float* rd = ABCD + (size_t)d * 512;
  const float* rs = ABCD + (size_t)s * 512;
  float z = rd[256 + c] + rs[384 + c] + Ee;
  float sig = 1.f / (1.f + __expf(-z));
  float msg = rd[c] + sig * rs[128 + c];
  unsafeAtomicAdd(&agg[(size_t)d * 128 + c], msg);
}

// ---- BatchNorm stats: per-channel sum & sumsq over N rows ----
__global__ __launch_bounds__(128) void bn_stats(const float* __restrict__ X,
                                                float* __restrict__ st) {
  int c = threadIdx.x;
  size_t r0 = (size_t)blockIdx.x * 512;
  float s = 0.f, ss = 0.f;
  for (int r = 0; r < 512; ++r) {
    float v = X[(r0 + r) * 128 + c];
    s += v; ss += v * v;
  }
  unsafeAtomicAdd(&st[c], s);
  unsafeAtomicAdd(&st[128 + c], ss);
}
__global__ __launch_bounds__(128) void bn_fin(const float* __restrict__ st,
                                              const float* __restrict__ g,
                                              const float* __restrict__ b,
                                              float* __restrict__ scsh) {
  int c = threadIdx.x;
  float m = st[c] * (1.f / 65536.f);
  float var = st[128 + c] * (1.f / 65536.f) - m * m;
  float inv = rsqrtf(var + 1e-5f);
  float sc = g[c] * inv;
  scsh[c] = sc;
  scsh[128 + c] = b[c] - m * sc;
}

// ---- elementwise: 0: bn(A)  1: relu(bn(A))+B  2: A+B ----
__global__ __launch_bounds__(256) void ew_kernel(const float* __restrict__ A,
                                                 const float* __restrict__ Badd,
                                                 const float* __restrict__ scsh,
                                                 float* __restrict__ out, int mode) {
  size_t i = (size_t)blockIdx.x * 256 + threadIdx.x;
  int c = (int)(i & 127);
  float a = A[i], v;
  if (mode == 0)      v = a * scsh[c] + scsh[128 + c];
  else if (mode == 1) v = fmaxf(a * scsh[c] + scsh[128 + c], 0.f) + Badd[i];
  else                v = a + Badd[i];
  out[i] = v;
}

// ---- dense MHA, one block per (graph, head): M=256, HD=32, full WMMA ----
__global__ __launch_bounds__(256) void attn_wmma(const float* __restrict__ qkv,
                                                 float* __restrict__ O) {
  __shared__ _Float16 Qs[256 * 32];    // [m][d]
  __shared__ _Float16 Ks[256 * 32];    // [m][d]
  __shared__ _Float16 Vt[32 * 256];    // transposed: [d][m]
  __shared__ _Float16 Ps[8][16 * 32];  // wave-private P staging
  int g = blockIdx.x, hh = blockIdx.y;
  int tid = threadIdx.x;
  const float* base = qkv + (size_t)g * 256 * 384 + hh * 32;
  for (int i = tid; i < 256 * 32; i += 256) {
    int m = i >> 5, d = i & 31;
    Qs[i] = (_Float16)base[(size_t)m * 384 + d];
    Ks[i] = (_Float16)base[(size_t)m * 384 + 128 + d];
    Vt[d * 256 + m] = (_Float16)base[(size_t)m * 384 + 256 + d];
  }
  __syncthreads();
  int wave = tid >> 5, lane = tid & 31;
  int hi = lane >> 4, r = lane & 15;
  const float scale = 0.17677669529663687f;  // 1/sqrt(32)
  for (int panel = 0; panel < 2; ++panel) {
    int sr = panel * 128 + wave * 16;  // 16-row strip per wave
    // S strip = Q[sr:sr+16,:] @ K^T : 16 tiles of 16x16, K=32 (one wmma each)
    v8f s[16];
    v16h aq = load_a_frag(Qs, 32, sr, 0, lane);
#pragma unroll
    for (int nt = 0; nt < 16; ++nt) {
      v16h b = load_b_frag(Ks, 32, nt * 16, 0, lane);
      v8f z = {};
      s[nt] = wmma16(aq, b, z);
    }
    // softmax: row = hi*8+j fixed per (lane-group, j); cols spread over
    // 16 tiles x 16 lanes -> in-lane reduce over nt, then shfl over lanes 0..15
#pragma unroll
    for (int j = 0; j < 8; ++j) {
      float rm = -3.0e38f;
#pragma unroll
      for (int nt = 0; nt < 16; ++nt) rm = fmaxf(rm, s[nt][j]);
#pragma unroll
      for (int m = 1; m <= 8; m <<= 1) rm = fmaxf(rm, __shfl_xor(rm, m, 32));
      float rs = 0.f;
#pragma unroll
      for (int nt = 0; nt < 16; ++nt) {
        float e = __expf((s[nt][j] - rm) * scale);
        s[nt][j] = e; rs += e;
      }
#pragma unroll
      for (int m = 1; m <= 8; m <<= 1) rs += __shfl_xor(rs, m, 32);
      float inv = 1.f / rs;
#pragma unroll
      for (int nt = 0; nt < 16; ++nt) s[nt][j] *= inv;
    }
    // O strip = P @ V : loop K panels of 32, restage P into A-layout via LDS
    v8f o0 = {}, o1 = {};
    _Float16* pw = &Ps[wave][0];
    for (int kt = 0; kt < 8; ++kt) {
#pragma unroll
      for (int t2 = 0; t2 < 2; ++t2) {
        int nt = kt * 2 + t2;
#pragma unroll
        for (int j = 0; j < 8; ++j)
          pw[(hi * 8 + j) * 32 + t2 * 16 + r] = (_Float16)s[nt][j];
      }
      v16h a  = load_a_frag(pw, 32, 0, 0, lane);
      v16h b0 = load_b_frag(Vt, 256, 0,  kt * 32, lane);
      v16h b1 = load_b_frag(Vt, 256, 16, kt * 32, lane);
      o0 = wmma16(a, b0, o0);
      o1 = wmma16(a, b1, o1);
    }
#pragma unroll
    for (int j = 0; j < 8; ++j) {
      size_t row = (size_t)g * 256 + sr + hi * 8 + j;
      O[row * 128 + hh * 32 + r]      = o0[j];
      O[row * 128 + hh * 32 + 16 + r] = o1[j];
    }
  }
}

// ---- mean-pool per graph + final linear ----
__global__ __launch_bounds__(128) void pool_kernel(const float* __restrict__ h,
                                                   const float* __restrict__ w,
                                                   const float* __restrict__ b,
                                                   float* __restrict__ out) {
  __shared__ float red[128];
  int g = blockIdx.x, c = threadIdx.x;
  float s = 0.f;
  for (int r = 0; r < 256; ++r) s += h[((size_t)g * 256 + r) * 128 + c];
  red[c] = s * (1.f / 256.f) * w[c];
  __syncthreads();
  for (int st = 64; st > 0; st >>= 1) {
    if (c < st) red[c] += red[c + st];
    __syncthreads();
  }
  if (c == 0) out[g] = red[0] + b[0];
}

// ---- tiny utility kernels ----
__global__ void fillzero(float* p, int n) {
  int i = blockIdx.x * 256 + threadIdx.x;
  if (i < n) p[i] = 0.f;
}
__global__ void cvt16(_Float16* dst, const float* src, int n) {
  int i = blockIdx.x * 256 + threadIdx.x;
  if (i < n) dst[i] = (_Float16)src[i];
}
__global__ void copyf(float* dst, const float* src, int n) {
  int i = blockIdx.x * 256 + threadIdx.x;
  if (i < n) dst[i] = src[i];
}

extern "C" void kernel_launch(void* const* d_in, const int* in_sizes, int n_in,
                              void* d_out, int out_size, void* d_ws, size_t ws_size,
                              hipStream_t stream) {
  const float* x        = (const float*)d_in[0];
  const float* x_pe     = (const float*)d_in[1];
  const float* ea       = (const float*)d_in[2];
  const int*   ei       = (const int*)d_in[3];
  const float* ne_w     = (const float*)d_in[5];
  const float* ne_b     = (const float*)d_in[6];
  const float* pe_w     = (const float*)d_in[7];
  const float* pe_b     = (const float*)d_in[8];
  const float* cA_w = (const float*)d_in[9];  const float* cA_b = (const float*)d_in[10];
  const float* cB_w = (const float*)d_in[11]; const float* cB_b = (const float*)d_in[12];
  const float* cC_w = (const float*)d_in[13]; const float* cC_b = (const float*)d_in[14];
  const float* cD_w = (const float*)d_in[15]; const float* cD_b = (const float*)d_in[16];
  const float* cE_w = (const float*)d_in[17]; const float* cE_b = (const float*)d_in[18];
  const float* cbn_g = (const float*)d_in[19]; const float* cbn_b = (const float*)d_in[20];
  const float* ai_w = (const float*)d_in[21]; const float* ai_b = (const float*)d_in[22];
  const float* ao_w = (const float*)d_in[23]; const float* ao_b = (const float*)d_in[24];
  const float* bn1_g = (const float*)d_in[25]; const float* bn1_b = (const float*)d_in[26];
  const float* bn2_g = (const float*)d_in[27]; const float* bn2_b = (const float*)d_in[28];
  const float* bn3_g = (const float*)d_in[29]; const float* bn3_b = (const float*)d_in[30];
  const float* m1_w = (const float*)d_in[31]; const float* m1_b = (const float*)d_in[32];
  const float* m2_w = (const float*)d_in[33]; const float* m2_b = (const float*)d_in[34];
  const float* lo_w = (const float*)d_in[35]; const float* lo_b = (const float*)d_in[36];
  float* out = (float*)d_out;

  // workspace layout
  char* W = (char*)d_ws;
  size_t off = 0;
  auto alloc = [&](size_t bytes) -> void* {
    void* p = W + off;
    off = (off + bytes + 255) & ~(size_t)255;
    return p;
  };
  _Float16* w4h  = (_Float16*)alloc((size_t)2 * 512 * 128 * 2);  // packed A|B|C|D
  _Float16* wqh  = (_Float16*)alloc((size_t)2 * 384 * 128 * 2);
  _Float16* woh  = (_Float16*)alloc((size_t)2 * 128 * 128 * 2);
  _Float16* wm1h = (_Float16*)alloc((size_t)2 * 256 * 128 * 2);
  _Float16* wm2h = (_Float16*)alloc((size_t)2 * 128 * 256 * 2);
  float* bias4 = (float*)alloc((size_t)2 * 512 * 4);
  float* stats = (float*)alloc((size_t)640 * 4);  // [0:256) sum/ssq, [256:512) scale/shift
  float* hB   = (float*)alloc((size_t)NN * 128 * 4);
  float* t512 = (float*)alloc((size_t)NN * 512 * 4);  // ABCD / qkv / ffpre
  float* aggB = (float*)alloc((size_t)NN * 128 * 4);
  float* s1B  = (float*)alloc((size_t)NN * 128 * 4);
  float* h1B  = (float*)alloc((size_t)NN * 128 * 4);
  float* oB   = (float*)alloc((size_t)NN * 128 * 4);
  float* scsh = stats + 256;

  const int CVT = 256;
  // weight conversion + bias packing
  for (int l = 0; l < 2; ++l) {
    const float* cw[4] = {cA_w, cB_w, cC_w, cD_w};
    const float* cb[4] = {cA_b, cB_b, cC_b, cD_b};
    for (int q = 0; q < 4; ++q) {
      cvt16<<<(16384 + CVT - 1) / CVT, CVT, 0, stream>>>(
          w4h + (size_t)l * 65536 + q * 16384, cw[q] + (size_t)l * 16384, 16384);
      copyf<<<1, 128, 0, stream>>>(bias4 + l * 512 + q * 128, cb[q] + l * 128, 128);
    }
    cvt16<<<(49152 + CVT - 1) / CVT, CVT, 0, stream>>>(wqh + (size_t)l * 49152, ai_w + (size_t)l * 49152, 49152);
    cvt16<<<(16384 + CVT - 1) / CVT, CVT, 0, stream>>>(woh + (size_t)l * 16384, ao_w + (size_t)l * 16384, 16384);
    cvt16<<<(32768 + CVT - 1) / CVT, CVT, 0, stream>>>(wm1h + (size_t)l * 32768, m1_w + (size_t)l * 32768, 32768);
    cvt16<<<(32768 + CVT - 1) / CVT, CVT, 0, stream>>>(wm2h + (size_t)l * 32768, m2_w + (size_t)l * 32768, 32768);
  }

  node_emb<<<NN / 2, 256, 0, stream>>>(x, x_pe, ne_w, ne_b, pe_w, pe_b, hB);

  const int EWG = (NN * 128) / 256;  // elementwise grid
  for (int l = 0; l < 2; ++l) {
    // ---- GatedGCN local conv ----
    gemm_wmma<<<dim3(NN / 32, 512 / 64), 128, 0, stream>>>(
        hB, w4h + (size_t)l * 65536, bias4 + l * 512, nullptr, t512, 128, 512, 0);
    fillzero<<<(NN * 128 + 255) / 256, 256, 0, stream>>>(aggB, NN * 128);
    edge_gatedgcn<<<ENL / 2, 256, 0, stream>>>(t512, ei, ea, cE_w + (size_t)l * 2048,
                                               cE_b + l * 128, aggB);
    fillzero<<<1, 256, 0, stream>>>(stats, 256);
    bn_stats<<<128, 128, 0, stream>>>(aggB, stats);
    bn_fin<<<1, 128, 0, stream>>>(stats, cbn_g + l * 128, cbn_b + l * 128, scsh);
    ew_kernel<<<EWG, 256, 0, stream>>>(aggB, hB, scsh, s1B, 1);  // s1 = relu(bn(agg)) + h
    fillzero<<<1, 256, 0, stream>>>(stats, 256);
    bn_stats<<<128, 128, 0, stream>>>(s1B, stats);
    bn_fin<<<1, 128, 0, stream>>>(stats, bn1_g + l * 128, bn1_b + l * 128, scsh);
    ew_kernel<<<EWG, 256, 0, stream>>>(s1B, nullptr, scsh, h1B, 0);  // h1 = bn(s1)

    // ---- global attention ----
    gemm_wmma<<<dim3(NN / 32, 384 / 64), 128, 0, stream>>>(
        hB, wqh + (size_t)l * 49152, ai_b + l * 384, nullptr, t512, 128, 384, 0);  // qkv
    attn_wmma<<<dim3(GG, 4), 256, 0, stream>>>(t512, oB);                          // o
    gemm_wmma<<<dim3(NN / 32, 128 / 64), 128, 0, stream>>>(
        oB, woh + (size_t)l * 16384, ao_b + l * 128, hB, s1B, 128, 128, 0);  // s2 = o@Wo+b + h
    fillzero<<<1, 256, 0, stream>>>(stats, 256);
    bn_stats<<<128, 128, 0, stream>>>(s1B, stats);
    bn_fin<<<1, 128, 0, stream>>>(stats, bn2_g + l * 128, bn2_b + l * 128, scsh);
    ew_kernel<<<EWG, 256, 0, stream>>>(s1B, nullptr, scsh, oB, 0);        // h2 = bn(s2)

    // ---- combine + FFN ----
    ew_kernel<<<EWG, 256, 0, stream>>>(h1B, oB, scsh, s1B, 2);            // out = h1 + h2
    gemm_wmma<<<dim3(NN / 32, 256 / 64), 128, 0, stream>>>(
        s1B, wm1h + (size_t)l * 32768, m1_b + l * 256, nullptr, t512, 128, 256, 1);
    gemm_wmma<<<dim3(NN / 32, 128 / 64), 128, 0, stream>>>(
        t512, wm2h + (size_t)l * 32768, m2_b + l * 128, s1B, h1B, 256, 128, 0);  // s3 = ff + out
    fillzero<<<1, 256, 0, stream>>>(stats, 256);
    bn_stats<<<128, 128, 0, stream>>>(h1B, stats);
    bn_fin<<<1, 128, 0, stream>>>(stats, bn3_g + l * 128, bn3_b + l * 128, scsh);
    ew_kernel<<<EWG, 256, 0, stream>>>(h1B, nullptr, scsh, hB, 0);        // h = bn(s3)
  }

  pool_kernel<<<GG, 128, 0, stream>>>(hB, lo_w, lo_b, out);
}
